// DSQGAttention_25237227831342
// MI455X (gfx1250) — matrix operand order repaired and checked by reading, not compiled, not run on hardware
//
#include <hip/hip_runtime.h>
#include <hip/hip_bf16.h>
#include <cstdint>

// ---------------------------------------------------------------------------
// Types for CDNA5 WMMA
// ---------------------------------------------------------------------------
typedef __attribute__((ext_vector_type(16))) __bf16 v16bf;
typedef __attribute__((ext_vector_type(8)))  float  v8f;

#define WMMA_BF16(A, B, C) \
    __builtin_amdgcn_wmma_f32_16x16x32_bf16(false, (A), false, (B), (short)0, (C), false, false)

// Problem constants
static constexpr int NSEQ = 4096;
static constexpr int D    = 1024;
static constexpr int H    = 16;
static constexpr int HD   = 64;
static constexpr int NOFF = 24;

__constant__ int c_offs[NOFF] = {0, 1, 2, 3, 4, 6, 8, 12, 16, 24, 32, 48, 64, 96, 128,
                                 192, 256, 384, 512, 768, 1024, 1536, 2048, 3072};

// ---------------------------------------------------------------------------
// Helpers
// ---------------------------------------------------------------------------
__device__ __forceinline__ v16bf load16bf(const __bf16* p0, const __bf16* p1) {
    union { uint4 u[2]; v16bf v; } u;
    u.u[0] = *reinterpret_cast<const uint4*>(p0);
    u.u[1] = *reinterpret_cast<const uint4*>(p1);
    return u.v;
}

__device__ __forceinline__ void split_bf16(float f, __bf16& hi, __bf16& lo) {
    hi = (__bf16)f;
    lo = (__bf16)(f - (float)hi);
}

// ---------------------------------------------------------------------------
// Prep kernels
// ---------------------------------------------------------------------------
// Elementwise split: X[f32] -> Xhi, Xlo [bf16], same layout
__global__ void k_split(const float* __restrict__ X,
                        __bf16* __restrict__ Xhi, __bf16* __restrict__ Xlo, int total) {
    int i = blockIdx.x * blockDim.x + threadIdx.x;
    if (i >= total) return;
    __bf16 h, l;
    split_bf16(X[i], h, l);
    Xhi[i] = h;
    Xlo[i] = l;
}

// Transpose + split: W [K,N] f32 (row-major) -> WT_hi/lo [N,K] bf16
__global__ void k_splitT(const float* __restrict__ W,
                         __bf16* __restrict__ Thi, __bf16* __restrict__ Tlo,
                         int K, int N) {
    int i = blockIdx.x * blockDim.x + threadIdx.x;
    if (i >= K * N) return;
    int k = i / N, n = i % N;                 // coalesced read
    __bf16 h, l;
    split_bf16(W[i], h, l);
    size_t o = (size_t)n * K + k;             // scattered one-time write
    Thi[o] = h;
    Tlo[o] = l;
}

// ---------------------------------------------------------------------------
// WMMA GEMM:  C[M,N] = A[M,K] @ B[K,N] + bias   (B given transposed: BT [N,K])
// A and B in bf16 hi/lo split form; fp32 accumulation via v_wmma_f32_16x16x32_bf16
// One wave computes a 16x64 tile (4 accumulators).
// MODE 0: C[m,n] = acc + bias[n]                           (write f32 to Cout)
// MODE 1: g = sigmoid(acc + bias[n]); val = g * mulIn[m,n];
//         split val -> outHi/outLo (feeds next GEMM)
// ---------------------------------------------------------------------------
template <int MODE>
__global__ __launch_bounds__(128)
void k_gemm(const __bf16* __restrict__ Ahi, const __bf16* __restrict__ Alo,
            const __bf16* __restrict__ BThi, const __bf16* __restrict__ BTlo,
            const float* __restrict__ bias,
            float* __restrict__ Cout,
            const float* __restrict__ mulIn,
            __bf16* __restrict__ outHi, __bf16* __restrict__ outLo,
            int M, int N, int K) {
    const int lane = threadIdx.x & 31;
    const int wid  = blockIdx.x * (blockDim.x >> 5) + (threadIdx.x >> 5);
    const int ngroups = N >> 6;                     // tiles of 64 columns
    const int mtile = wid / ngroups;
    const int ng    = wid % ngroups;
    if (mtile * 16 >= M) return;

    // A operand: 16-bit A 16x32 layout. Lane l<16 -> row l, K = {kc..kc+7, kc+16..kc+23}
    //            Lane l>=16 -> row l-16, K = {kc+8..kc+15, kc+24..kc+31}
    const int mrow  = mtile * 16 + (lane & 15);
    const int aksel = (lane < 16) ? 0 : 8;
    const __bf16* aH = Ahi + (size_t)mrow * K + aksel;
    const __bf16* aL = Alo + (size_t)mrow * K + aksel;

    // B operand: 16-bit B 32x16 layout. Lane l<16 -> col l, K = {kc..kc+7, kc+16..kc+23}
    //            Lane l>=16 -> col l-16, K = {kc+8..kc+15, kc+24..kc+31}
    const int ncol0 = ng * 64 + (lane & 15);
    const int bksel = (lane < 16) ? 0 : 8;

    v8f acc0 = {}, acc1 = {}, acc2 = {}, acc3 = {};

    for (int kc = 0; kc < K; kc += 32) {
        v16bf ah = load16bf(aH + kc, aH + kc + 16);
        v16bf al = load16bf(aL + kc, aL + kc + 16);

#pragma unroll
        for (int nt = 0; nt < 4; ++nt) {
            const size_t boff = (size_t)(ncol0 + nt * 16) * K + bksel + kc;
            v16bf bh = load16bf(BThi + boff, BThi + boff + 16);
            v16bf bl = load16bf(BTlo + boff, BTlo + boff + 16);
            v8f a = (nt == 0) ? acc0 : (nt == 1) ? acc1 : (nt == 2) ? acc2 : acc3;
            a = WMMA_BF16(ah, bh, a);   // hi*hi
            a = WMMA_BF16(ah, bl, a);   // hi*lo
            a = WMMA_BF16(al, bh, a);   // lo*hi   (lo*lo dropped: ~2^-16 rel)
            if (nt == 0) acc0 = a; else if (nt == 1) acc1 = a;
            else if (nt == 2) acc2 = a; else acc3 = a;
        }
    }

    // Epilogue. C/D layout: VGPR r, lane l -> row mtile*16 + r + (l<16?0:8), col l&15
    const int rbase = mtile * 16 + ((lane < 16) ? 0 : 8);
#pragma unroll
    for (int nt = 0; nt < 4; ++nt) {
        const v8f a = (nt == 0) ? acc0 : (nt == 1) ? acc1 : (nt == 2) ? acc2 : acc3;
        const int n = ng * 64 + nt * 16 + (lane & 15);
        const float bn = bias[n];
#pragma unroll
        for (int r = 0; r < 8; ++r) {
            const int m = rbase + r;
            const size_t o = (size_t)m * N + n;
            const float v = a[r] + bn;
            if constexpr (MODE == 0) {
                Cout[o] = v;
            } else {
                const float g   = 1.0f / (1.0f + __expf(-v));
                const float val = g * mulIn[o];
                __bf16 h, l;
                split_bf16(val, h, l);
                outHi[o] = h;
                outLo[o] = l;
            }
        }
    }
}

// ---------------------------------------------------------------------------
// 24-tap local attention. One wave32 per (t, head); 2 dims/lane of HD=64.
// Exactly mirrors reference masking: invalid taps get score = pos_bias (they
// stay in the softmax denominator) but contribute zero V.
// ---------------------------------------------------------------------------
__global__ __launch_bounds__(256)
void k_attn(const float* __restrict__ qkv,      // [N, 3D], q|k|v
            const float* __restrict__ pos_bias, // [NOFF, H]
            float* __restrict__ out) {          // [N, D]
    const int lane = threadIdx.x & 31;
    const int wid  = blockIdx.x * (blockDim.x >> 5) + (threadIdx.x >> 5);
    const int h = wid & (H - 1);
    const int t = wid >> 4;
    if (t >= NSEQ) return;

    const float scale = 0.125f;  // HD^-0.5
    const float* Q = qkv + (size_t)t * (3 * D) + h * HD;
    const float q0 = Q[lane], q1 = Q[lane + 32];

    float sc[NOFF];
#pragma unroll
    for (int o = 0; o < NOFF; ++o) {
        const int src = t - c_offs[o];
        float s = 0.0f;
        if (src >= 0) {
            const float* Kp = qkv + (size_t)src * (3 * D) + D + h * HD;
            s = q0 * Kp[lane] + q1 * Kp[lane + 32];
        }
#pragma unroll
        for (int m = 16; m >= 1; m >>= 1) s += __shfl_xor(s, m, 32);
        sc[o] = s * scale + pos_bias[o * H + h];
    }

    float mx = sc[0];
#pragma unroll
    for (int o = 1; o < NOFF; ++o) mx = fmaxf(mx, sc[o]);
    float e[NOFF], den = 0.0f;
#pragma unroll
    for (int o = 0; o < NOFF; ++o) { e[o] = __expf(sc[o] - mx); den += e[o]; }
    const float inv = 1.0f / den;

    float o0 = 0.0f, o1 = 0.0f;
#pragma unroll
    for (int o = 0; o < NOFF; ++o) {
        const int src = t - c_offs[o];
        if (src >= 0) {
            const float* Vp = qkv + (size_t)src * (3 * D) + 2 * D + h * HD;
            const float w = e[o] * inv;
            o0 += w * Vp[lane];
            o1 += w * Vp[lane + 32];
        }
    }
    out[(size_t)t * D + h * HD + lane]      = o0;
    out[(size_t)t * D + h * HD + lane + 32] = o1;
}

// ---------------------------------------------------------------------------
// Host side
// ---------------------------------------------------------------------------
extern "C" void kernel_launch(void* const* d_in, const int* in_sizes, int n_in,
                              void* d_out, int out_size, void* d_ws, size_t ws_size,
                              hipStream_t stream) {
    const float* x     = (const float*)d_in[0];   // [4096,1024]
    const float* Wqkv  = (const float*)d_in[1];   // [1024,3072]
    const float* bqkv  = (const float*)d_in[2];   // [3072]
    const float* Wout  = (const float*)d_in[3];   // [1024,1024]
    const float* bout  = (const float*)d_in[4];   // [1024]
    const float* Wgate = (const float*)d_in[5];   // [1024,1024]
    const float* bgate = (const float*)d_in[6];   // [1024]
    const float* pbias = (const float*)d_in[7];   // [24,16]

    char* ws = (char*)d_ws;
    const size_t MiB = 1024ull * 1024ull;
    __bf16* xhi  = (__bf16*)(ws + 0 * MiB);       // 8 MiB
    __bf16* xlo  = (__bf16*)(ws + 8 * MiB);       // 8 MiB
    float*  qkvb = (float*)(ws + 16 * MiB);       // 48 MiB
    float*  aout = (float*)(ws + 64 * MiB);       // 16 MiB
    __bf16* wqh  = (__bf16*)(ws + 80 * MiB);      // 6 MiB
    __bf16* wql  = (__bf16*)(ws + 86 * MiB);      // 6 MiB
    __bf16* wgh  = (__bf16*)(ws + 92 * MiB);      // 2 MiB
    __bf16* wgl  = (__bf16*)(ws + 94 * MiB);      // 2 MiB
    __bf16* woh  = (__bf16*)(ws + 96 * MiB);      // 2 MiB
    __bf16* wol  = (__bf16*)(ws + 98 * MiB);      // 2 MiB
    __bf16* a2h  = (__bf16*)(ws + 100 * MiB);     // 8 MiB
    __bf16* a2l  = (__bf16*)(ws + 108 * MiB);     // 8 MiB

    // --- prep: bf16 hi/lo splits (x) and transposed splits (weights) ---
    {
        int tot = NSEQ * D;
        k_split<<<(tot + 255) / 256, 256, 0, stream>>>(x, xhi, xlo, tot);
    }
    k_splitT<<<(D * 3 * D + 255) / 256, 256, 0, stream>>>(Wqkv,  wqh, wql, D, 3 * D);
    k_splitT<<<(D * D + 255) / 256, 256, 0, stream>>>(Wgate, wgh, wgl, D, D);
    k_splitT<<<(D * D + 255) / 256, 256, 0, stream>>>(Wout,  woh, wol, D, D);

    // --- qkv = x @ Wqkv + bqkv ---
    {
        int waves = (NSEQ / 16) * (3 * D / 64);   // 12288
        k_gemm<0><<<waves / 4, 128, 0, stream>>>(xhi, xlo, wqh, wql, bqkv,
                                                 qkvb, nullptr, nullptr, nullptr,
                                                 NSEQ, 3 * D, D);
    }

    // --- 24-tap attention ---
    {
        int waves = NSEQ * H;                     // 65536
        k_attn<<<waves / 8, 256, 0, stream>>>(qkvb, pbias, aout);
    }

    // --- a2 = attn_out * sigmoid(x @ Wgate + bgate), split to bf16 hi/lo ---
    {
        int waves = (NSEQ / 16) * (D / 64);       // 4096
        k_gemm<1><<<waves / 4, 128, 0, stream>>>(xhi, xlo, wgh, wgl, bgate,
                                                 nullptr, aout, a2h, a2l,
                                                 NSEQ, D, D);
    }

    // --- out = a2 @ Wout + bout ---
    {
        int waves = (NSEQ / 16) * (D / 64);       // 4096
        k_gemm<0><<<waves / 4, 128, 0, stream>>>(a2h, a2l, woh, wol, bout,
                                                 (float*)d_out, nullptr, nullptr, nullptr,
                                                 NSEQ, D, D);
    }
}
